// GCN_33535104647614
// MI455X (gfx1250) — matrix operand order repaired and checked by reading, not compiled, not run on hardware
//
#include <hip/hip_runtime.h>
#include <hip/hip_bf16.h>
#include <math.h>

// ---------------------------------------------------------------------------
// Types for CDNA5 WMMA (wave32): v_wmma_f32_16x16x32_bf16
// ---------------------------------------------------------------------------
typedef __attribute__((ext_vector_type(16))) __bf16 v16bf;
typedef __attribute__((ext_vector_type(8)))  float  v8f;

#define NFEAT 512
#define NCODE 128
#define NHID  128
#define NCLS  40

// ---------------------------------------------------------------------------
// Pack weight matrix B[K, M] (fp32) into bf16 WMMA B-fragment layout:
// for k-step kt and 16-col tile t, 32 lanes x v16bf (32B/lane), where lane l
// holds column (t*16 + (l&15)) and element j = K row (kt*32 + 16*(l>>4) + j).
// One wave per (kt, t). B is tiny (<=256KB) and L2-hot; runs once per GEMM.
// ---------------------------------------------------------------------------
__global__ void __launch_bounds__(32)
pack_B_frag(const float* __restrict__ B, __bf16* __restrict__ Bpk,
            int M, int nct)
{
    const int lane = threadIdx.x;
    const int kt   = blockIdx.x;
    const int t    = blockIdx.y;
    const int half = lane >> 4;
    const int col  = t * 16 + (lane & 15);

    v16bf frag;
#pragma unroll
    for (int j = 0; j < 16; ++j) {
        int   k = kt * 32 + 16 * half + j;
        float v = (col < M) ? B[(size_t)k * M + col] : 0.0f;
        frag[j] = (__bf16)v;
    }
    ((v16bf*)Bpk)[((size_t)kt * nct + t) * 32 + lane] = frag;
}

// ---------------------------------------------------------------------------
// GEMM: C[N, M] = A[N, K] @ Bpk (+ bias), fp32 A/C, bf16 WMMA compute.
// One wave computes a 16 x (16*NT) slab of C.
// Per 32-wide k-step:
//   - A tile (16x32 fp32) loaded coalesced (16 batched b32 loads/lane),
//     converted to bf16, staged through LDS into a contiguous-k A-fragment layout (k = 16*half + j, as in Bpk),
//     read back as one v16bf (2x ds_load_b128).
//   - All NT pre-packed B fragments are loaded up-front into distinct
//     registers (one clause of 2*NT global_load_b128) so the WMMA chain
//     only pays incremental loadcnt waits.
// ---------------------------------------------------------------------------
template <int NT>
__global__ void __launch_bounds__(32)
gemm_wmma_bf16(const float* __restrict__ A,
               const __bf16* __restrict__ Bpk,
               const float* __restrict__ bias,
               float* __restrict__ C,
               int N, int K, int M, int ldc, int nct)
{
    __shared__ v16bf ldsAv[32];            // 1 KB, fragment-layout A tile
    __bf16* lA = (__bf16*)ldsAv;

    const int lane    = threadIdx.x;                 // 0..31
    const int rowBase = blockIdx.x * 16;
    const int t0      = blockIdx.y * NT;             // first 16-col tile
    const int half    = lane >> 4;                   // 0 | 1
    const int q       = lane & 15;
    const v16bf* Bv   = (const v16bf*)Bpk;

    v8f acc[NT];
#pragma unroll
    for (int t = 0; t < NT; ++t) {
        int n = (t0 + t) * 16 + q;
        float bv = (bias != nullptr && n < M) ? bias[n] : 0.0f;
#pragma unroll
        for (int v = 0; v < 8; ++v) acc[t][v] = bv;
    }

    const int KT = K >> 5;                           // k-steps
    for (int kt = 0; kt < KT; ++kt) {
        // ---- Stage A: lane loads column (kt*32+lane) of rows 0..15
        // (coalesced 128B per row); all 16 loads batched, then converted and
        // scattered into fragment layout:
        //   element (r, k): frag lane = r + 16*(k>>4), j = k&15
        const int k  = lane;
        const int lt = 16 * (k >> 4);
        const int j  = k & 15;
        float areg[16];
#pragma unroll
        for (int r = 0; r < 16; ++r) {
            int rr = rowBase + r; rr = (rr < N) ? rr : (N - 1);
            areg[r] = A[(size_t)rr * K + (kt * 32 + k)];
        }
#pragma unroll
        for (int r = 0; r < 16; ++r) {
            lA[(lt + r) * 16 + j] = (__bf16)areg[r];
        }
        __syncthreads();

        // ---- Preload all B fragments (distinct regs -> one load clause),
        // then the A fragment, then issue the WMMA chain.
        v16bf bfr[NT];
#pragma unroll
        for (int t = 0; t < NT; ++t) {
            bfr[t] = Bv[((size_t)kt * nct + t0 + t) * 32 + lane];
        }
        const v16bf afrag = ldsAv[lane];             // 2x ds_load_b128
#pragma unroll
        for (int t = 0; t < NT; ++t) {
            acc[t] = __builtin_amdgcn_wmma_f32_16x16x32_bf16(
                false, afrag, false, bfr[t], (short)0, acc[t], false, false);
        }
        __syncthreads();
    }

    // C/D layout: VGPR v -> row (v + 8*half), col = lane&15.
#pragma unroll
    for (int t = 0; t < NT; ++t) {
        int n = (t0 + t) * 16 + q;
        if (n < M) {
#pragma unroll
            for (int v = 0; v < 8; ++v) {
                int row = rowBase + v + 8 * half;
                if (row < N) C[(size_t)row * ldc + n] = acc[t][v];
            }
        }
    }
}

// ---------------------------------------------------------------------------
// SpMM scatter: Out[dst[e], ofs:ofs+D] += vals[e] * X[src[e], 0:D]
// One wave per edge, float4 (b128) gathers, fp32 global atomics for the
// segment sum. 8 edges per 256-thread block. D4 = D/4.
// ---------------------------------------------------------------------------
template <int D4>
__global__ void __launch_bounds__(256)
spmm_scatter(const float* __restrict__ vals,
             const int*   __restrict__ src,
             const int*   __restrict__ dst,
             const float* __restrict__ X, int ldx,
             float* __restrict__ Out, int ldo, int ofs, int E)
{
    const int lane = threadIdx.x & 31;
    const int e    = blockIdx.x * 8 + (threadIdx.x >> 5);
    if (e >= E) return;
    const float   v  = vals[e];
    const float4* xr = (const float4*)(X + (size_t)src[e] * ldx);
    float*        o  = Out + (size_t)dst[e] * ldo + ofs;
#pragma unroll
    for (int i = lane; i < D4; i += 32) {
        float4 x4 = xr[i];
        atomicAdd(&o[4 * i + 0], v * x4.x);
        atomicAdd(&o[4 * i + 1], v * x4.y);
        atomicAdd(&o[4 * i + 2], v * x4.z);
        atomicAdd(&o[4 * i + 3], v * x4.w);
    }
}

// ---------------------------------------------------------------------------
// Zero fp32 buffer (atomic accumulators must restart at 0 every call).
// ---------------------------------------------------------------------------
__global__ void zero_f32(float* __restrict__ p, size_t n)
{
    size_t i      = (size_t)blockIdx.x * blockDim.x + threadIdx.x;
    size_t stride = (size_t)gridDim.x * blockDim.x;
    for (; i < n; i += stride) p[i] = 0.0f;
}

// ---------------------------------------------------------------------------
// Fused bias + ELU(alpha=0.1) over the [N,256] concat buffer:
// cols 0..127 get gc2_b, cols 128..255 get gc1_b.
// ---------------------------------------------------------------------------
__global__ void bias_elu_concat(float* __restrict__ z,
                                const float* __restrict__ b2,
                                const float* __restrict__ b1,
                                size_t total)
{
    size_t i      = (size_t)blockIdx.x * blockDim.x + threadIdx.x;
    size_t stride = (size_t)gridDim.x * blockDim.x;
    for (; i < total; i += stride) {
        int   c = (int)(i & 255u);
        float b = (c < NHID) ? b2[c] : b1[c - NHID];
        float v = z[i] + b;
        z[i] = (v > 0.0f) ? v : 0.1f * expm1f(v);
    }
}

// ---------------------------------------------------------------------------
// Fused bias + log_softmax over C=40 classes, one thread per node.
// ---------------------------------------------------------------------------
__global__ void bias_logsoftmax(const float* __restrict__ logits,
                                const float* __restrict__ bias,
                                float* __restrict__ out,
                                int N, int C)
{
    int i = blockIdx.x * blockDim.x + threadIdx.x;
    if (i >= N) return;
    const float* r = logits + (size_t)i * C;
    float mx = -INFINITY;
    for (int c = 0; c < C; ++c) mx = fmaxf(mx, r[c] + bias[c]);
    float s = 0.0f;
    for (int c = 0; c < C; ++c) s += expf(r[c] + bias[c] - mx);
    float lse = logf(s) + mx;
    float* o = out + (size_t)i * C;
    for (int c = 0; c < C; ++c) o[c] = r[c] + bias[c] - lse;
}

// ---------------------------------------------------------------------------
// Orchestration
// ---------------------------------------------------------------------------
extern "C" void kernel_launch(void* const* d_in, const int* in_sizes, int n_in,
                              void* d_out, int out_size, void* d_ws, size_t ws_size,
                              hipStream_t stream)
{
    (void)n_in; (void)out_size; (void)ws_size;

    const float* x_in  = (const float*)d_in[0];
    const int*   src   = (const int*)  d_in[1];
    const int*   dst   = (const int*)  d_in[2];
    const float* avals = (const float*)d_in[3];
    const float* enc_W = (const float*)d_in[4];
    const float* enc_b = (const float*)d_in[5];
    const float* dec_W = (const float*)d_in[6];
    const float* dec_b = (const float*)d_in[7];
    const float* gc1_W = (const float*)d_in[8];
    const float* gc1_b = (const float*)d_in[9];
    const float* gc2_W = (const float*)d_in[10];
    const float* gc2_b = (const float*)d_in[11];
    const float* gc3_W = (const float*)d_in[12];
    const float* gc3_b = (const float*)d_in[13];

    const int N = in_sizes[0] / NFEAT;   // 50000
    const int E = in_sizes[1];           // 600000

    float* out = (float*)d_out;
    float* x_out     = out;                       // [N, 40]
    float* x_recover = out + (size_t)N * NCLS;    // [N, 512]

    // Workspace layout (fp32, then bf16 packed-weight scratch)
    float* ws      = (float*)d_ws;
    float* x_code  = ws;                               // [N,128]
    float* t2      = x_code + (size_t)N * NCODE;       // [N,128]  x_in  @ gc2_W
    float* t1      = t2     + (size_t)N * NHID;        // [N,128]  x_code@ gc1_W
    float* z       = t1     + (size_t)N * NHID;        // [N,256]  concat(h2,h1)
    float* y3      = z      + (size_t)N * (2*NHID);    // [N,40]   z @ gc3_W
    float* logits  = y3     + (size_t)N * NCLS;        // [N,40]   spmm out
    __bf16* Bpk    = (__bf16*)(logits + (size_t)N * NCLS); // <=128KB packed B

    const int rowTiles = (N + 15) / 16;
    dim3 blk32(32);

    // ---- 1) x_code = x_in @ enc_W + enc_b        [N,512]x[512,128]
    pack_B_frag<<<dim3(NFEAT / 32, NCODE / 16), blk32, 0, stream>>>(
        enc_W, Bpk, NCODE, NCODE / 16);
    gemm_wmma_bf16<8><<<dim3(rowTiles, 1), blk32, 0, stream>>>(
        x_in, Bpk, enc_b, x_code, N, NFEAT, NCODE, NCODE, NCODE / 16);

    // ---- 2) t2 = x_in @ gc2_W                    [N,512]x[512,128]
    pack_B_frag<<<dim3(NFEAT / 32, NHID / 16), blk32, 0, stream>>>(
        gc2_W, Bpk, NHID, NHID / 16);
    gemm_wmma_bf16<8><<<dim3(rowTiles, 1), blk32, 0, stream>>>(
        x_in, Bpk, nullptr, t2, N, NFEAT, NHID, NHID, NHID / 16);

    // ---- 3) t1 = x_code @ gc1_W                  [N,128]x[128,128]
    pack_B_frag<<<dim3(NCODE / 32, NHID / 16), blk32, 0, stream>>>(
        gc1_W, Bpk, NHID, NHID / 16);
    gemm_wmma_bf16<8><<<dim3(rowTiles, 1), blk32, 0, stream>>>(
        x_code, Bpk, nullptr, t1, N, NCODE, NHID, NHID, NHID / 16);

    // ---- 4) x_recover = x_code @ dec_W + dec_b -> directly into d_out slice
    pack_B_frag<<<dim3(NCODE / 32, NFEAT / 16), blk32, 0, stream>>>(
        dec_W, Bpk, NFEAT, NFEAT / 16);
    gemm_wmma_bf16<8><<<dim3(rowTiles, 4), blk32, 0, stream>>>(
        x_code, Bpk, dec_b, x_recover, N, NCODE, NFEAT, NFEAT, NFEAT / 16);

    // ---- 5) zero concat accumulator z [N,256]
    zero_f32<<<2048, 256, 0, stream>>>(z, (size_t)N * (2 * NHID));

    // ---- 6) h2 -> z[:, 0:128]   (spmm of t2)
    spmm_scatter<NHID / 4><<<(E + 7) / 8, 256, 0, stream>>>(
        avals, src, dst, t2, NHID, z, 2 * NHID, 0, E);
    // ---- 7) h1 -> z[:, 128:256] (spmm of t1)
    spmm_scatter<NHID / 4><<<(E + 7) / 8, 256, 0, stream>>>(
        avals, src, dst, t1, NHID, z, 2 * NHID, NHID, E);

    // ---- 8) z = elu(z + concat(gc2_b, gc1_b))
    bias_elu_concat<<<2048, 256, 0, stream>>>(z, gc2_b, gc1_b,
                                              (size_t)N * (2 * NHID));

    // ---- 9) y3 = z @ gc3_W                       [N,256]x[256,40] (3 tiles)
    pack_B_frag<<<dim3((2 * NHID) / 32, 3), blk32, 0, stream>>>(
        gc3_W, Bpk, NCLS, 3);
    gemm_wmma_bf16<3><<<dim3(rowTiles, 1), blk32, 0, stream>>>(
        z, Bpk, nullptr, y3, N, 2 * NHID, NCLS, NCLS, 3);

    // ---- 10) zero logits, 11) spmm(y3) -> logits
    zero_f32<<<1024, 256, 0, stream>>>(logits, (size_t)N * NCLS);
    spmm_scatter<NCLS / 4><<<(E + 7) / 8, 256, 0, stream>>>(
        avals, src, dst, y3, NCLS, logits, NCLS, 0, E);

    // ---- 12) x_out = log_softmax(logits + gc3_b)
    bias_logsoftmax<<<(N + 127) / 128, 128, 0, stream>>>(logits, gc3_b,
                                                         x_out, N, NCLS);
}
